// LSTMPredictor_68289980006995
// MI455X (gfx1250) — compile-verified, matrix operand order checked by reading, not run on hardware
//
#include <hip/hip_runtime.h>
#include <hip/hip_bf16.h>

// ---------------------------------------------------------------------------
// 2-layer LSTM (B=256, T=512, F=64, H=512) + linear head, for MI455X (gfx1250)
// bf16 WMMA (v_wmma_f32_16x16x32_bf16), fp32 accumulation.
// Each 16x16 tile of [batch x H] (all 4 gates) is computed by FOUR waves
// splitting the K range (software-pipelined fragment loads); partials are
// combined through LDS. 256 blocks/step to fill the WGP array.
// ---------------------------------------------------------------------------

#define LSTM_B   256
#define LSTM_T   512
#define LSTM_F   64
#define LSTM_H   512
#define LSTM_G   2048   // 4*H

typedef __attribute__((ext_vector_type(16))) __bf16 v16bf;
typedef __attribute__((ext_vector_type(8)))  __bf16 v8bf;
typedef __attribute__((ext_vector_type(8)))  float  v8f;

// Per-lane bf16 fragment: two contiguous 16B chunks at p and p+16
// (ISA 16-bit A/B layout; the lane's k-half offset is applied by the caller).
__device__ __forceinline__ v16bf load_frag16(const __bf16* __restrict__ p) {
    union { v16bf v; v8bf h[2]; } u;
    u.h[0] = *(const v8bf*)(p);
    u.h[1] = *(const v8bf*)(p + 16);
    return u.v;
}

__device__ __forceinline__ float fast_sigmoid(float x) {
    return 1.0f / (1.0f + __expf(-x));
}

#define WMMA_BF16(A, Bm, Cm) \
    __builtin_amdgcn_wmma_f32_16x16x32_bf16(false, (A), false, (Bm), (short)0, (Cm), false, false)

// Strided-chunk GEMM segment with software pipelining.
// Processes chunks c = ks, ks+4, ... < NC (chunk = 32 K-elements).
// NC is compile-time so the rotated double-buffer fully unrolls.
template<int NC>
__device__ __forceinline__ void gemm_seg(
    const __bf16* __restrict__ a,
    const __bf16* __restrict__ b0, const __bf16* __restrict__ b1,
    const __bf16* __restrict__ b2, const __bf16* __restrict__ b3,
    int ks, v8f acc[4])
{
    int c = ks;
    if (c >= NC) return;
    v16bf aC = load_frag16(a  + c * 32);
    v16bf B0 = load_frag16(b0 + c * 32);
    v16bf B1 = load_frag16(b1 + c * 32);
    v16bf B2 = load_frag16(b2 + c * 32);
    v16bf B3 = load_frag16(b3 + c * 32);
    #pragma unroll
    for (int i = 1; i < (NC + 3) / 4; ++i) {
        const int cn = ks + i * 4;
        if (cn < NC) {
            // issue next chunk's loads before consuming current fragments
            v16bf aN = load_frag16(a  + cn * 32);
            v16bf N0 = load_frag16(b0 + cn * 32);
            v16bf N1 = load_frag16(b1 + cn * 32);
            v16bf N2 = load_frag16(b2 + cn * 32);
            v16bf N3 = load_frag16(b3 + cn * 32);
            acc[0] = WMMA_BF16(aC, B0, acc[0]);
            acc[1] = WMMA_BF16(aC, B1, acc[1]);
            acc[2] = WMMA_BF16(aC, B2, acc[2]);
            acc[3] = WMMA_BF16(aC, B3, acc[3]);
            aC = aN; B0 = N0; B1 = N1; B2 = N2; B3 = N3;
        }
    }
    acc[0] = WMMA_BF16(aC, B0, acc[0]);
    acc[1] = WMMA_BF16(aC, B1, acc[1]);
    acc[2] = WMMA_BF16(aC, B2, acc[2]);
    acc[3] = WMMA_BF16(aC, B3, acc[3]);
}

// One LSTM timestep for one layer.
//   gates[b, 4H] = Hprev @ Whh^T + A1 @ W1^T + bias
//   c = sig(f)*c + sig(i)*tanh(g);  h = sig(o)*tanh(c)
// Grid: 256 blocks x 256 threads. Block = 8 waves = 2 tiles x 4 K-split waves.
// 512 tiles = 16 (M=batch) x 32 (N of H).
template<int K1, int LDA1>
__global__ __launch_bounds__(256) void lstm_step_kernel(
    const __bf16* __restrict__ A1,
    const __bf16* __restrict__ W1,     // [2048, K1]
    const __bf16* __restrict__ Hprev,  // [256, 512]
    const __bf16* __restrict__ Whh,    // [2048, 512]
    const float*  __restrict__ bias,   // [2048] = b_ih + b_hh
    float*        __restrict__ C,      // [256, 512] fp32, in place
    __bf16*       __restrict__ Hout)   // [256, 512]
{
    __shared__ float lds_acc[32][2][3][32];   // [j=g*8+v][tileInBlock][ks-1][lane]

    const int lane = threadIdx.x & 31;
    const int wave = threadIdx.x >> 5;
    const int tib  = wave >> 2;               // tile in block: 0..1
    const int ks   = wave & 3;                // K-split quarter: 0..3
    const int tile = blockIdx.x * 2 + tib;    // 0..511
    const int m_base = (tile >> 5) * 16;      // batch tile
    const int n_base = (tile & 31) * 16;      // H-column tile

    const int row16 = lane & 15;              // M (A) / N (B)
    const int khalf = (lane >> 4) * 8;        // per-lane k-half (ISA layout)

    v8f acc[4];
    #pragma unroll
    for (int g = 0; g < 4; ++g) acc[g] = (v8f){0.f,0.f,0.f,0.f,0.f,0.f,0.f,0.f};

    // ---- recurrent part: 16 chunks, this wave takes c = ks, ks+4, ... ----
    gemm_seg<LSTM_H / 32>(
        Hprev + (m_base + row16) * LSTM_H + khalf,
        Whh + (0 * LSTM_H + n_base + row16) * LSTM_H + khalf,
        Whh + (1 * LSTM_H + n_base + row16) * LSTM_H + khalf,
        Whh + (2 * LSTM_H + n_base + row16) * LSTM_H + khalf,
        Whh + (3 * LSTM_H + n_base + row16) * LSTM_H + khalf,
        ks, acc);

    // ---- input part: K1/32 chunks, same interleave ----
    gemm_seg<K1 / 32>(
        A1 + (m_base + row16) * LDA1 + khalf,
        W1 + (0 * LSTM_H + n_base + row16) * K1 + khalf,
        W1 + (1 * LSTM_H + n_base + row16) * K1 + khalf,
        W1 + (2 * LSTM_H + n_base + row16) * K1 + khalf,
        W1 + (3 * LSTM_H + n_base + row16) * K1 + khalf,
        ks, acc);

    // ---- combine K-split partials through LDS ----
    if (ks != 0) {
        #pragma unroll
        for (int g = 0; g < 4; ++g)
            #pragma unroll
            for (int v = 0; v < 8; ++v)
                lds_acc[g * 8 + v][tib][ks - 1][lane] = acc[g][v];  // conflict-free
    }
    __syncthreads();
    if (ks != 0) return;

    #pragma unroll
    for (int g = 0; g < 4; ++g)
        #pragma unroll
        for (int v = 0; v < 8; ++v) {
            float s = acc[g][v];
            s += lds_acc[g * 8 + v][tib][0][lane];
            s += lds_acc[g * 8 + v][tib][1][lane];
            s += lds_acc[g * 8 + v][tib][2][lane];
            acc[g][v] = s;
        }

    // ---- epilogue: gate nonlinearities + c/h update ----
    // C/D f32 16x16 layout: element (VGPR v, lane L) -> M = v + 8*(L/16), N = L%16
    const int n    = n_base + row16;
    const int mofs = (lane >> 4) * 8;
    const float bi = bias[0 * LSTM_H + n];
    const float bf = bias[1 * LSTM_H + n];
    const float bg = bias[2 * LSTM_H + n];
    const float bo = bias[3 * LSTM_H + n];
    #pragma unroll
    for (int v = 0; v < 8; ++v) {
        const int m = m_base + mofs + v;
        const float ig = fast_sigmoid(acc[0][v] + bi);
        const float fg = fast_sigmoid(acc[1][v] + bf);
        const float gg = tanhf(acc[2][v] + bg);
        const float og = fast_sigmoid(acc[3][v] + bo);
        const int idx = m * LSTM_H + n;
        const float c_new = fg * C[idx] + ig * gg;
        C[idx] = c_new;
        Hout[idx] = (__bf16)(og * tanhf(c_new));
    }
}

__global__ void f32_to_bf16_kernel(const float* __restrict__ in,
                                   __bf16* __restrict__ out, int n) {
    int i = blockIdx.x * blockDim.x + threadIdx.x;
    if (i < n) out[i] = (__bf16)in[i];
}

__global__ void bias_combine_kernel(const float* __restrict__ a,
                                    const float* __restrict__ b,
                                    float* __restrict__ o, int n) {
    int i = blockIdx.x * blockDim.x + threadIdx.x;
    if (i < n) o[i] = a[i] + b[i];
}

__global__ void final_linear_kernel(const __bf16* __restrict__ h,   // [256,512]
                                    const float* __restrict__ Wlin, // [512]
                                    const float* __restrict__ blin, // [1]
                                    float* __restrict__ out) {      // [256]
    int b = blockIdx.x * blockDim.x + threadIdx.x;
    if (b < LSTM_B) {
        float s = blin[0];
        #pragma unroll 8
        for (int k = 0; k < LSTM_H; ++k) s += (float)h[b * LSTM_H + k] * Wlin[k];
        out[b] = s;
    }
}

extern "C" void kernel_launch(void* const* d_in, const int* in_sizes, int n_in,
                              void* d_out, int out_size, void* d_ws, size_t ws_size,
                              hipStream_t stream) {
    (void)in_sizes; (void)n_in; (void)out_size; (void)ws_size;

    const float* x     = (const float*)d_in[0];   // [B,T,F]
    const float* Wih0  = (const float*)d_in[1];   // [4H,F]
    const float* Whh0  = (const float*)d_in[2];   // [4H,H]
    const float* bih0  = (const float*)d_in[3];
    const float* bhh0  = (const float*)d_in[4];
    const float* Wih1  = (const float*)d_in[5];   // [4H,H]
    const float* Whh1  = (const float*)d_in[6];   // [4H,H]
    const float* bih1  = (const float*)d_in[7];
    const float* bhh1  = (const float*)d_in[8];
    const float* Wlin  = (const float*)d_in[9];   // [1,H]
    const float* blin  = (const float*)d_in[10];  // [1]
    float* out = (float*)d_out;

    // ---- workspace layout (256B aligned slices) ----
    char* ws = (char*)d_ws;
    size_t off = 0;
    auto alloc = [&](size_t bytes) {
        void* p = ws + off;
        off += (bytes + 255) & ~(size_t)255;
        return p;
    };
    __bf16* x_bf    = (__bf16*)alloc((size_t)LSTM_B * LSTM_T * LSTM_F * 2);
    __bf16* Wih0_bf = (__bf16*)alloc((size_t)LSTM_G * LSTM_F * 2);
    __bf16* Whh0_bf = (__bf16*)alloc((size_t)LSTM_G * LSTM_H * 2);
    __bf16* Wih1_bf = (__bf16*)alloc((size_t)LSTM_G * LSTM_H * 2);
    __bf16* Whh1_bf = (__bf16*)alloc((size_t)LSTM_G * LSTM_H * 2);
    float*  bias0   = (float*)alloc((size_t)LSTM_G * 4);
    float*  bias1   = (float*)alloc((size_t)LSTM_G * 4);
    __bf16* h0buf[2]; __bf16* h1buf[2];
    h0buf[0] = (__bf16*)alloc((size_t)LSTM_B * LSTM_H * 2);
    h0buf[1] = (__bf16*)alloc((size_t)LSTM_B * LSTM_H * 2);
    h1buf[0] = (__bf16*)alloc((size_t)LSTM_B * LSTM_H * 2);
    h1buf[1] = (__bf16*)alloc((size_t)LSTM_B * LSTM_H * 2);
    float* c0 = (float*)alloc((size_t)LSTM_B * LSTM_H * 4);
    float* c1 = (float*)alloc((size_t)LSTM_B * LSTM_H * 4);

    // ---- zero-init recurrent state (deterministic every call) ----
    hipMemsetAsync(h0buf[0], 0, (size_t)LSTM_B * LSTM_H * 2, stream);
    hipMemsetAsync(h0buf[1], 0, (size_t)LSTM_B * LSTM_H * 2, stream);
    hipMemsetAsync(h1buf[0], 0, (size_t)LSTM_B * LSTM_H * 2, stream);
    hipMemsetAsync(h1buf[1], 0, (size_t)LSTM_B * LSTM_H * 2, stream);
    hipMemsetAsync(c0, 0, (size_t)LSTM_B * LSTM_H * 4, stream);
    hipMemsetAsync(c1, 0, (size_t)LSTM_B * LSTM_H * 4, stream);

    // ---- precision conversion + bias combine ----
    auto cvt = [&](const float* src, __bf16* dst, int n) {
        f32_to_bf16_kernel<<<(n + 255) / 256, 256, 0, stream>>>(src, dst, n);
    };
    cvt(x,    x_bf,    LSTM_B * LSTM_T * LSTM_F);
    cvt(Wih0, Wih0_bf, LSTM_G * LSTM_F);
    cvt(Whh0, Whh0_bf, LSTM_G * LSTM_H);
    cvt(Wih1, Wih1_bf, LSTM_G * LSTM_H);
    cvt(Whh1, Whh1_bf, LSTM_G * LSTM_H);
    bias_combine_kernel<<<(LSTM_G + 255) / 256, 256, 0, stream>>>(bih0, bhh0, bias0, LSTM_G);
    bias_combine_kernel<<<(LSTM_G + 255) / 256, 256, 0, stream>>>(bih1, bhh1, bias1, LSTM_G);

    // ---- sequential recurrence: 512 steps x 2 fused layers ----
    // 512 tiles x 4 K-split waves @ 8 waves/block -> 256 blocks per step.
    for (int t = 0; t < LSTM_T; ++t) {
        const __bf16* h0p = h0buf[t & 1];
        __bf16*       h0n = h0buf[(t + 1) & 1];
        const __bf16* h1p = h1buf[t & 1];
        __bf16*       h1n = h1buf[(t + 1) & 1];
        // layer 0: input part = x[:, t, :], lda = T*F, K1 = F
        lstm_step_kernel<LSTM_F, LSTM_T * LSTM_F><<<256, 256, 0, stream>>>(
            x_bf + (size_t)t * LSTM_F,
            Wih0_bf, h0p, Whh0_bf, bias0, c0, h0n);
        // layer 1: input part = h0 of this step, lda = H, K1 = H
        lstm_step_kernel<LSTM_H, LSTM_H><<<256, 256, 0, stream>>>(
            h0n,
            Wih1_bf, h1p, Whh1_bf, bias1, c1, h1n);
    }

    // t = 511 wrote h1buf[0]
    final_linear_kernel<<<1, 256, 0, stream>>>(h1buf[0], Wlin, blin, out);
}
